// ProbabalisticSlotAttention_50216757625628
// MI455X (gfx1250) — matrix operand order repaired and checked by reading, not compiled, not run on hardware
//
#include <hip/hip_runtime.h>
#include <hip/hip_bf16.h>

// ---------------------------------------------------------------------------
// Probabilistic Slot Attention for MI455X (gfx1250), wave32 + WMMA bf16 + TDM.
// B=16, N=1024, D=512, K=8, 3 iterations.
// ---------------------------------------------------------------------------

#define B_  16
#define N_  1024
#define D_  512
#define K_  8
#define EPS_     1e-8f
#define LN_EPS_  1e-5f
#define LOG2PI_  1.8378770664093453f

typedef __bf16 bf16;
typedef __attribute__((ext_vector_type(16))) __bf16 v16bf;
typedef __attribute__((ext_vector_type(8)))  float  v8f;
typedef unsigned int __attribute__((ext_vector_type(4))) v4u;
typedef int          __attribute__((ext_vector_type(4))) v4i;
typedef int          __attribute__((ext_vector_type(8))) v8i;

// Tensor Data Mover availability (ROCm7.2 clang-22: 5-arg; clang-23: 6-arg).
#if defined(__has_builtin)
#  if __has_builtin(__builtin_amdgcn_tensor_load_to_lds) && \
      __has_builtin(__builtin_amdgcn_s_wait_tensorcnt)
#    define HAVE_TDM 1
#  endif
#endif
#ifndef HAVE_TDM
#  define HAVE_TDM 0
#endif

// ---------------------------------------------------------------------------
// 1) Convert Wk / Wv to bf16 *transposed* ([D_out][D_in]) so the GEMM's
//    B-tile can be staged n-major in LDS -> per-lane WMMA B fragments become
//    two contiguous 16B runs (ds_load_b128) instead of 16 scalar u16 loads.
// ---------------------------------------------------------------------------
__global__ void cvt_w_kernel(const float* __restrict__ Wk,
                             const float* __restrict__ Wv,
                             bf16* __restrict__ wkbT, bf16* __restrict__ wvbT) {
    int idx = blockIdx.x * 256 + threadIdx.x;
    if (idx < D_ * D_) {
        int k = idx >> 9, n = idx & (D_ - 1);
        wkbT[n * D_ + k] = (bf16)Wk[idx];
    } else if (idx < 2 * D_ * D_) {
        int i = idx - D_ * D_;
        int k = i >> 9, n = i & (D_ - 1);
        wvbT[n * D_ + k] = (bf16)Wv[i];
    }
}

// ---------------------------------------------------------------------------
// 2) LayerNorm over D, emit bf16 activations for the WMMA GEMM.
// ---------------------------------------------------------------------------
__global__ void ln_kernel(const float* __restrict__ emb,
                          const float* __restrict__ g,
                          const float* __restrict__ be,
                          bf16* __restrict__ xb) {
    int row = blockIdx.x;
    int tid = threadIdx.x;
    const float* r = emb + (size_t)row * D_;
    float v0 = r[tid], v1 = r[tid + 256];

    __shared__ float rs[256], rss[256];
    rs[tid]  = v0 + v1;
    rss[tid] = v0 * v0 + v1 * v1;
    __syncthreads();
    for (int o = 128; o > 0; o >>= 1) {
        if (tid < o) { rs[tid] += rs[tid + o]; rss[tid] += rss[tid + o]; }
        __syncthreads();
    }
    __shared__ float mv[2];
    if (tid == 0) {
        float m   = rs[0] * (1.0f / D_);
        float var = rss[0] * (1.0f / D_) - m * m;
        mv[0] = m; mv[1] = rsqrtf(var + LN_EPS_);
    }
    __syncthreads();
    float m = mv[0], rstd = mv[1];
    bf16* o = xb + (size_t)row * D_;
    o[tid]       = (bf16)((v0 - m) * rstd * g[tid]       + be[tid]);
    o[tid + 256] = (bf16)((v1 - m) * rstd * g[tid + 256] + be[tid + 256]);
}

// ---------------------------------------------------------------------------
// 3) Slot sampling + sigma init + loop-invariant queries + mixing init.
// ---------------------------------------------------------------------------
__global__ void init_kernel(const float* __restrict__ slots_mu,
                            const float* __restrict__ slots_ls,
                            const float* __restrict__ noise_init,
                            const float* __restrict__ mix_c,
                            const float* __restrict__ Wq,
                            const float* __restrict__ bq,
                            float* __restrict__ sigma,
                            float* __restrict__ queries,
                            float* __restrict__ mixing) {
    int bk = blockIdx.x;            // b*K + k
    int k  = bk & (K_ - 1);
    int tid = threadIdx.x;
    __shared__ float srow[D_];
    #pragma unroll
    for (int h = 0; h < 2; ++h) {
        int d = tid + h * 256;
        float sg = __expf(slots_ls[k * D_ + d]);
        sigma[(size_t)bk * D_ + d] = sg;
        srow[d] = slots_mu[k * D_ + d] + sg * noise_init[(size_t)bk * D_ + d];
    }
    __syncthreads();
    #pragma unroll
    for (int h = 0; h < 2; ++h) {
        int dout = tid + h * 256;
        float acc = bq[dout];
        for (int d = 0; d < D_; ++d) acc += srow[d] * Wq[d * D_ + dout];
        queries[(size_t)bk * D_ + dout] = acc;
    }
    if (tid == 0) mixing[bk] = mix_c[k];
}

// ---------------------------------------------------------------------------
// 4) WMMA bf16 GEMM: out[16384 x 512] = x_bf16 @ W_bf16 + bias (f32 out).
//    Block tile 128(M) x 64(N); 8 waves in 4x2, each wave 32x32 = 2x2 WMMA
//    tiles; K-loop in steps of 32 through LDS.
//    A tile is DMA'd via the Tensor Data Mover (TENSOR_LOAD_TO_LDS) by wave 0
//    when available; B tile staged from the pre-transposed weights.
// ---------------------------------------------------------------------------
__global__ void __launch_bounds__(256)
proj_gemm_kernel(const bf16* __restrict__ xb,
                 const bf16* __restrict__ wkbT, const bf16* __restrict__ wvbT,
                 const float* __restrict__ bk, const float* __restrict__ bv,
                 float* __restrict__ keys, float* __restrict__ values) {
    const bf16*  WT   = blockIdx.z == 0 ? wkbT : wvbT;
    const float* bias = blockIdx.z == 0 ? bk  : bv;
    float*       out  = blockIdx.z == 0 ? keys : values;

    const int n0   = blockIdx.x * 64;
    const int row0 = blockIdx.y * 128;
    const int tid  = threadIdx.x;
    const int wave = tid >> 5, lane = tid & 31;
    const int wm = wave >> 1, wn = wave & 1;       // 4x2 wave grid
    const int m_base = wm * 32, n_base = wn * 32;
    const int lr = lane & 15, lh = lane >> 4;

    __shared__ bf16 sA[128][40];   // 128 rows x 32 k (+8 pad), 16B-aligned rows
    __shared__ bf16 sBt[64][40];   // 64 cols (n) x 32 k (+8 pad): n-major!

    v8f acc[2][2] = {};

    // B staging: thread -> (n row of sBt, 8-k segment)
    const int bn = tid >> 2, bseg = tid & 3;
#if !(HAVE_TDM && defined(__gfx1250__))
    const int ar = tid >> 1, ah = tid & 1;         // A staging fallback
#endif

    for (int step = 0; step < D_ / 32; ++step) {
        const int kk = step * 32;
        // ---- B tile 64(n) x 32(k) bf16 from transposed weights (16B/thr) ----
        const float4* gb = reinterpret_cast<const float4*>(
            WT + (size_t)(n0 + bn) * D_ + kk + bseg * 8);
        float4 b0 = gb[0];
#if !(HAVE_TDM && defined(__gfx1250__))
        const float4* ga = reinterpret_cast<const float4*>(
            xb + (size_t)(row0 + ar) * D_ + kk + ah * 16);
        float4 a0 = ga[0], a1 = ga[1];
#endif
        if (step + 1 < D_ / 32) {   // prefetch next tiles -> global_prefetch_b8
            __builtin_prefetch(WT + (size_t)(n0 + bn) * D_ + kk + 32 + bseg * 8, 0, 1);
            __builtin_prefetch(xb + (size_t)(row0 + (tid >> 1)) * D_ + kk + 32, 0, 1);
        }
        __syncthreads();            // previous compute done before overwrite

#if HAVE_TDM && defined(__gfx1250__)
        if (tid < 32) {
            // Tensor DMA descriptor (ISA 8.3/8.4): 2-D tile 32(k) x 128(rows),
            // data_size=2B, tensor_dim0_stride = D_ elements.
            unsigned long long ga64 = (unsigned long long)(size_t)
                (xb + (size_t)row0 * D_ + kk);
            unsigned lds_off = (unsigned)(size_t)&sA[0][0];
            v4u g0;
            g0[0] = 1u;                                   // count=1, user mode
            g0[1] = lds_off;                              // lds_addr
            g0[2] = (unsigned)ga64;                       // global_addr[31:0]
            g0[3] = (unsigned)(ga64 >> 32) | (2u << 30);  // addr[56:32] | type=2
            v8i g1;
            g1[0] = (int)(1u << 16);                      // data_size = 2 bytes
            g1[1] = (int)((unsigned)(D_ & 0xFFFF) << 16); // tensor_dim0[15:0]
            g1[2] = (int)((unsigned)((B_ * N_) & 0xFFFF) << 16); // tensor_dim1 lo
            g1[3] = (int)(32u << 16);                     // tile_dim0 = 32
            g1[4] = 128;                                  // tile_dim1 = 128
            g1[5] = D_;                                   // tensor_dim0_stride lo
            g1[6] = 0;
            g1[7] = 0;
            v4i gz = (v4i)0;
#if __clang_major__ >= 23
            __builtin_amdgcn_tensor_load_to_lds(g0, g1, gz, gz, (v8i)0, 0);
#else
            __builtin_amdgcn_tensor_load_to_lds(g0, g1, gz, gz, 0);
#endif
            __builtin_amdgcn_s_wait_tensorcnt(0);
        }
#else
        *reinterpret_cast<float4*>(&sA[ar][ah * 16])     = a0;
        *reinterpret_cast<float4*>(&sA[ar][ah * 16 + 8]) = a1;
#endif
        *reinterpret_cast<float4*>(&sBt[bn][bseg * 8]) = b0;
        __syncthreads();

        // ---- gather fragments per ISA 7.12.2 layout tables ----
        // A: lane holds row m, two contiguous 8-element k-runs -> 2x b128.
        // B (n-major LDS): lane holds col n, two contiguous 8-element k-runs.
        v16bf afrag[2], bfrag[2];
        #pragma unroll
        for (int mt = 0; mt < 2; ++mt) {
            int m = m_base + mt * 16 + lr;
            #pragma unroll
            for (int i = 0; i < 8; ++i) {
                int k0 = (i < 4 ? 2 * i : 16 + 2 * (i - 4)) + 8 * lh;
                afrag[mt][2 * i]     = sA[m][k0];
                afrag[mt][2 * i + 1] = sA[m][k0 + 1];
            }
        }
        #pragma unroll
        for (int nt = 0; nt < 2; ++nt) {
            int n = n_base + nt * 16 + lr;
            #pragma unroll
            for (int i = 0; i < 8; ++i) {
                int k0 = (i < 4 ? 2 * i : 16 + 2 * (i - 4)) + 8 * lh;
                bfrag[nt][2 * i]     = sBt[n][k0];
                bfrag[nt][2 * i + 1] = sBt[n][k0 + 1];
            }
        }
        #pragma unroll
        for (int mt = 0; mt < 2; ++mt)
            #pragma unroll
            for (int nt = 0; nt < 2; ++nt)
                acc[mt][nt] = __builtin_amdgcn_wmma_f32_16x16x32_bf16(
                    false, afrag[mt], false, bfrag[nt],
                    (short)0, acc[mt][nt], false, false);
    }

    // ---- epilogue: C layout (lane = col, VGPR j -> M = j + 8*lh) + bias ----
    #pragma unroll
    for (int mt = 0; mt < 2; ++mt) {
        #pragma unroll
        for (int nt = 0; nt < 2; ++nt) {
            int n = n0 + n_base + nt * 16 + lr;
            float bsn = bias[n];
            #pragma unroll
            for (int j = 0; j < 8; ++j) {
                int m = row0 + m_base + mt * 16 + j + 8 * lh;
                out[(size_t)m * D_ + n] = acc[mt][nt][j] + bsn;
            }
        }
    }
}

// ---------------------------------------------------------------------------
// 5) Per-iteration prep: w = 1/(sigma^2+eps), qw = q*w, and per-(b,k) constant
//    konst = -0.5*D*log(2pi) - 0.5*sum(log|sigma|+eps) - 0.5*sum(q^2 w).
//    Also zeroes the attention column sums. One block per (b,k).
// ---------------------------------------------------------------------------
__global__ void prep_kernel(const float* __restrict__ sigma,
                            const float* __restrict__ queries,
                            float* __restrict__ wbuf, float* __restrict__ qwbuf,
                            float* __restrict__ konst, float* __restrict__ colsum) {
    int bk = blockIdx.x, tid = threadIdx.x;
    float qq = 0.f, ld = 0.f;
    #pragma unroll
    for (int h = 0; h < 2; ++h) {
        size_t i = (size_t)bk * D_ + tid + h * 256;
        float s = sigma[i];
        float w = 1.0f / (s * s + EPS_);
        float q = queries[i];
        wbuf[i]  = w;
        qwbuf[i] = q * w;
        qq += q * q * w;
        ld += logf(fabsf(s) + EPS_);
    }
    __shared__ float r1[256], r2[256];
    r1[tid] = qq; r2[tid] = ld;
    __syncthreads();
    for (int o = 128; o > 0; o >>= 1) {
        if (tid < o) { r1[tid] += r1[tid + o]; r2[tid] += r2[tid + o]; }
        __syncthreads();
    }
    if (tid == 0) {
        konst[bk] = -0.5f * D_ * LOG2PI_ - 0.5f * r2[0] - 0.5f * r1[0];
        colsum[bk] = 0.f;
    }
}

// ---------------------------------------------------------------------------
// 6) Attention: gll -> mixing*gll -> k-normalize. One wave32 per row n,
//    8 rows per block; w/qw cached in LDS (batch-local). Column sums via
//    LDS atomics + one global atomic per k.
// ---------------------------------------------------------------------------
__global__ void __launch_bounds__(256)
attn_kernel(const float* __restrict__ keys,
            const float* __restrict__ wbuf, const float* __restrict__ qwbuf,
            const float* __restrict__ konst, const float* __restrict__ mixing,
            float* __restrict__ attn1, float* __restrict__ colsum) {
    const int b = blockIdx.x, tid = threadIdx.x;
    __shared__ float s_w[K_ * D_], s_qw[K_ * D_];
    __shared__ float cs[K_];
    #pragma unroll
    for (int i = 0; i < (K_ * D_) / 256; ++i) {
        int idx = tid + i * 256;
        s_w[idx]  = wbuf[(size_t)b * K_ * D_ + idx];
        s_qw[idx] = qwbuf[(size_t)b * K_ * D_ + idx];
    }
    if (tid < K_) cs[tid] = 0.f;
    __syncthreads();

    const int wave = tid >> 5, lane = tid & 31;
    const int n = blockIdx.y * 8 + wave;
    const float* krow = keys + ((size_t)b * N_ + n) * D_;

    float acc[K_] = {};
    #pragma unroll 4
    for (int j = 0; j < D_ / 32; ++j) {
        int d = lane + j * 32;
        float ky = krow[d];
        #pragma unroll
        for (int k = 0; k < K_; ++k)
            acc[k] += ky * (s_qw[k * D_ + d] - 0.5f * ky * s_w[k * D_ + d]);
    }
    #pragma unroll
    for (int k = 0; k < K_; ++k)
        #pragma unroll
        for (int off = 16; off >= 1; off >>= 1)
            acc[k] += __shfl_xor(acc[k], off, 32);

    float a[K_], denom = 0.f;
    #pragma unroll
    for (int k = 0; k < K_; ++k) {
        float gll = konst[b * K_ + k] + acc[k];
        a[k] = mixing[b * K_ + k] * gll;
        denom += a[k];
    }
    float inv = 1.0f / denom, myv = 0.f;
    #pragma unroll
    for (int k = 0; k < K_; ++k) if (lane == k) myv = a[k] * inv;
    if (lane < K_) {
        attn1[((size_t)b * N_ + n) * K_ + lane] = myv;
        atomicAdd(&cs[lane], myv);
    }
    __syncthreads();
    if (tid < K_) atomicAdd(&colsum[b * K_ + tid], cs[tid]);
}

// ---------------------------------------------------------------------------
// 7) Statistics: mu = attn2^T @ v, E2 = attn2^T @ v^2 (streaming over n),
//    sigma_new = E2 - mu^2*(2 - t_k), mixing = t_k/N.  attn2 = attn1/(S+eps).
// ---------------------------------------------------------------------------
__global__ void __launch_bounds__(256)
stats_kernel(const float* __restrict__ values, const float* __restrict__ attn1,
             const float* __restrict__ colsum,
             float* __restrict__ mu_buf, float* __restrict__ sigma,
             float* __restrict__ mixing) {
    const int b = blockIdx.y, tid = threadIdx.x;
    const int d = blockIdx.x * 256 + tid;
    __shared__ float sa[256];   // 32 rows x 8 k
    float macc[K_] = {}, eacc[K_] = {};
    for (int n0 = 0; n0 < N_; n0 += 32) {
        __syncthreads();
        sa[tid] = attn1[((size_t)b * N_ + n0 + (tid >> 3)) * K_ + (tid & 7)];
        __syncthreads();
        #pragma unroll 8
        for (int r = 0; r < 32; ++r) {
            float v = values[((size_t)b * N_ + n0 + r) * D_ + d];
            #pragma unroll
            for (int k = 0; k < K_; ++k) {
                float aa = sa[r * K_ + k];
                macc[k] += aa * v;
                eacc[k] += aa * v * v;
            }
        }
    }
    #pragma unroll
    for (int k = 0; k < K_; ++k) {
        float S = colsum[b * K_ + k];
        float inv = 1.0f / (S + EPS_);
        float m  = macc[k] * inv;
        float e2 = eacc[k] * inv;
        float t  = S * inv;
        mu_buf[((size_t)b * K_ + k) * D_ + d] = m;
        sigma[((size_t)b * K_ + k) * D_ + d]  = e2 - m * m * (2.0f - t);
    }
    if (blockIdx.x == 0 && tid < K_) {
        float S = colsum[b * K_ + tid];
        mixing[b * K_ + tid] = (S / (S + EPS_)) * (1.0f / N_);
    }
}

// ---------------------------------------------------------------------------
// 8) Final slots: slots = mu + max(|sigma|,eps)*noise_final.
// ---------------------------------------------------------------------------
__global__ void finalize_slots_kernel(const float* __restrict__ mu_buf,
                                      const float* __restrict__ sigma,
                                      const float* __restrict__ noise_final,
                                      float* __restrict__ out) {
    int bk = blockIdx.x, tid = threadIdx.x;
    #pragma unroll
    for (int h = 0; h < 2; ++h) {
        size_t i = (size_t)bk * D_ + tid + h * 256;
        float s = fmaxf(fabsf(sigma[i]), EPS_);
        out[i] = mu_buf[i] + s * noise_final[i];
    }
}

// ---------------------------------------------------------------------------
// 9) Output attn transposed: out[b,k,n] = attn1[b,n,k] / (S_k + eps).
// ---------------------------------------------------------------------------
__global__ void write_attn_kernel(const float* __restrict__ attn1,
                                  const float* __restrict__ colsum,
                                  float* __restrict__ out_attn) {
    int gid = blockIdx.x * 256 + threadIdx.x;   // over B*K*N
    int b = gid / (K_ * N_);
    int rem = gid - b * (K_ * N_);
    int k = rem / N_;
    int n = rem - k * N_;
    float S = colsum[b * K_ + k];
    out_attn[gid] = attn1[((size_t)b * N_ + n) * K_ + k] / (S + EPS_);
}

// ---------------------------------------------------------------------------
// Host launcher.
// Inputs (order): embeddings, noise_init, noise_final, slots_mu,
// slots_logsigma, mixing_coeffs, Wk, bk, Wq, bq, Wv, bv, ln_g, ln_b.
// Output: slots [B,K,D] then attnT [B,K,N], f32.
// ---------------------------------------------------------------------------
extern "C" void kernel_launch(void* const* d_in, const int* in_sizes, int n_in,
                              void* d_out, int out_size, void* d_ws, size_t ws_size,
                              hipStream_t stream) {
    const float* emb   = (const float*)d_in[0];
    const float* n_ini = (const float*)d_in[1];
    const float* n_fin = (const float*)d_in[2];
    const float* s_mu  = (const float*)d_in[3];
    const float* s_ls  = (const float*)d_in[4];
    const float* mix_c = (const float*)d_in[5];
    const float* Wk    = (const float*)d_in[6];
    const float* bk    = (const float*)d_in[7];
    const float* Wq    = (const float*)d_in[8];
    const float* bq    = (const float*)d_in[9];
    const float* Wv    = (const float*)d_in[10];
    const float* bv    = (const float*)d_in[11];
    const float* ln_g  = (const float*)d_in[12];
    const float* ln_b  = (const float*)d_in[13];
    float* out = (float*)d_out;

    char* ws = (char*)d_ws;
    // workspace layout (all offsets 256B-aligned), total ~83 MB
    bf16*  xb      = (bf16*) (ws + 0);           // 16 MB  [B*N, D] bf16
    bf16*  wkbT    = (bf16*) (ws + 16777216);    // 512 KB (transposed)
    bf16*  wvbT    = (bf16*) (ws + 17301504);    // 512 KB (transposed)
    float* keys    = (float*)(ws + 17825792);    // 32 MB
    float* values  = (float*)(ws + 51380224);    // 32 MB
    float* queries = (float*)(ws + 84934656);    // 256 KB
    float* sigma   = (float*)(ws + 85196800);    // 256 KB
    float* wbuf    = (float*)(ws + 85458944);    // 256 KB
    float* qwbuf   = (float*)(ws + 85721088);    // 256 KB
    float* mu_buf  = (float*)(ws + 85983232);    // 256 KB
    float* attn1   = (float*)(ws + 86245376);    // 512 KB [B,N,K]
    float* konst   = (float*)(ws + 86769664);    // 512 B
    float* colsum  = (float*)(ws + 86770176);    // 512 B
    float* mixing  = (float*)(ws + 86770688);    // 512 B

    cvt_w_kernel<<<(2 * D_ * D_) / 256, 256, 0, stream>>>(Wk, Wv, wkbT, wvbT);
    ln_kernel<<<B_ * N_, 256, 0, stream>>>(emb, ln_g, ln_b, xb);
    init_kernel<<<B_ * K_, 256, 0, stream>>>(s_mu, s_ls, n_ini, mix_c, Wq, bq,
                                             sigma, queries, mixing);
    proj_gemm_kernel<<<dim3(D_ / 64, (B_ * N_) / 128, 2), 256, 0, stream>>>(
        xb, wkbT, wvbT, bk, bv, keys, values);

    for (int it = 0; it < 3; ++it) {
        prep_kernel<<<B_ * K_, 256, 0, stream>>>(sigma, queries, wbuf, qwbuf,
                                                 konst, colsum);
        attn_kernel<<<dim3(B_, N_ / 8), 256, 0, stream>>>(keys, wbuf, qwbuf,
                                                          konst, mixing,
                                                          attn1, colsum);
        stats_kernel<<<dim3(D_ / 256, B_), 256, 0, stream>>>(values, attn1,
                                                             colsum, mu_buf,
                                                             sigma, mixing);
    }

    finalize_slots_kernel<<<B_ * K_, 256, 0, stream>>>(mu_buf, sigma, n_fin, out);
    write_attn_kernel<<<(B_ * K_ * N_) / 256, 256, 0, stream>>>(
        attn1, colsum, out + (size_t)B_ * K_ * D_);
}
